// GCNWithAttentionTwoHead_13469017441155
// MI455X (gfx1250) — matrix operand order repaired and checked by reading, not compiled, not run on hardware
//
#include <hip/hip_runtime.h>
#include <hip/hip_bf16.h>

// ---------------------------------------------------------------------------
// GCN-with-attention scoring on MI455X (gfx1250), wave32 + WMMA f16->f32.
// Pipeline:
//   K0 zero_pair : zero pairwise (N*N) + self_w (N) region of d_out
//   K1 cvt_t     : f32 weights -> transposed f16 weights in d_ws
//   K2 self_mlp  : g_i = MLP3(x_i; sW*) via v_wmma_f32_16x16x32_f16
//   K3 pair_mlp  : s_ij = MLP3([x_i, x_j]; nW*) via WMMA, 2 nodes / WG
//   K4 finalize  : wave-level softmax(b*|s|), pw, scatter, Y_pred
// ---------------------------------------------------------------------------

typedef __attribute__((ext_vector_type(16))) _Float16 v16h;
typedef __attribute__((ext_vector_type(8)))  _Float16 v8h;
typedef __attribute__((ext_vector_type(4)))  _Float16 v4h;
typedef __attribute__((ext_vector_type(8)))  float    v8f;

union F16x16 { v16h v; v8h h[2]; };

constexpr int NN   = 10000;
constexpr int KNBR = 32;
constexpr int DD   = 128;
constexpr int HH   = 256;
constexpr int LDA  = 264;   // 256 + 8 halfs pad: row stride 528B -> bank rotation 4
constexpr int LDB  = 72;    // 64  + 8 halfs pad: row stride 144B -> bank rotation 36
constexpr int MBLK = 64;    // rows per workgroup in the GEMM kernels

// f16 transposed-weight layout inside d_ws (offsets in halfs)
constexpr size_t OFF_N1 = 0;        // [256 out][256 in]
constexpr size_t OFF_N2 = 65536;    // [256][256]
constexpr size_t OFF_N3 = 131072;   // [256]
constexpr size_t OFF_S1 = 131328;   // [256 out][128 in]
constexpr size_t OFF_S2 = 164096;   // [256][256]
constexpr size_t OFF_S3 = 229632;   // [256]
constexpr size_t HALF_BYTES = 229888ull * 2ull;   // 459,776 B (16B aligned)

constexpr size_t SMEM_BYTES = (size_t)(MBLK * LDA + 256 * LDB) * 2 + MBLK * 4;

// ---------------------------------------------------------------------------
// Fragment loaders (CDNA5 WMMA f16 VGPR layouts, wave32)
// ---------------------------------------------------------------------------
__device__ __forceinline__ v16h load_a_frag(const _Float16* As, int m0, int k0, int lane) {
  // A 16x32: lane -> M = lane&15 ; kb = k0 + 8*(lane>=16)
  // halves 0..7  -> K = kb..kb+7   (VGPR0-3)
  // halves 8..15 -> K = kb+16..kb+23 (VGPR4-7)
  const _Float16* p = As + (size_t)(m0 + (lane & 15)) * LDA + k0 + ((lane >> 4) << 3);
  F16x16 f;
  f.h[0] = *(const v8h*)(p);
  f.h[1] = *(const v8h*)(p + 16);
  return f.v;
}

__device__ __forceinline__ v16h load_b_frag(const _Float16* Wb, int n0, int kt, int lane) {
  // B 32x16 from transposed weights Wb[n][k]: lane -> N = n0+(lane&15),
  // koff = kt + 8*(lane>=16); halves 0..7 -> K = koff..koff+7, halves 8..15 -> K = koff+16..koff+23
  const _Float16* p = Wb + (size_t)(n0 + (lane & 15)) * LDB + kt + ((lane >> 4) << 3);
  F16x16 f;
  f.h[0] = *(const v8h*)(p);
  f.h[1] = *(const v8h*)(p + 16);
  return f.v;
}

__device__ __forceinline__ v8f zero8() {
  v8f z = {0.f, 0.f, 0.f, 0.f, 0.f, 0.f, 0.f, 0.f};
  return z;
}

// One dense layer: acc[8] += As(64 x KDIM) @ Wt(KDIM x 256) for this wave's
// 16-row M-tile (m0) and 128-col N-half (wn). Weights streamed through LDS
// in 64-wide K chunks so each WG reads each weight element once per layer.
template<int KDIM>
__device__ __forceinline__ void gemm_layer(const _Float16* __restrict__ Wt,
                                           const _Float16* As, _Float16* Wb,
                                           int m0, int wn, int lane, int tid,
                                           v8f acc[8]) {
  for (int kc = 0; kc < KDIM; kc += 64) {
    // stage Wb[n=tid][0..63] <- Wt[tid][kc..kc+63]
    #pragma unroll
    for (int u = 0; u < 8; ++u)
      *(v8h*)(Wb + (size_t)tid * LDB + u * 8) =
          *(const v8h*)(Wt + (size_t)tid * KDIM + kc + u * 8);
    __syncthreads();
    #pragma unroll
    for (int kt = 0; kt < 64; kt += 32) {
      v16h a = load_a_frag(As, m0, kc + kt, lane);
      #pragma unroll
      for (int t = 0; t < 8; ++t) {
        v16h b = load_b_frag(Wb, wn * 128 + t * 16, kt, lane);
        acc[t] = __builtin_amdgcn_wmma_f32_16x16x32_f16(
            false, a, false, b, (short)0, acc[t], false, false);
      }
    }
    __syncthreads();
  }
}

// ---------------------------------------------------------------------------
// K0: zero pairwise matrix + self_w region
// ---------------------------------------------------------------------------
__global__ void zero_pair(float4* __restrict__ p, long long n4) {
  long long i = (long long)blockIdx.x * blockDim.x + threadIdx.x;
  long long stride = (long long)gridDim.x * blockDim.x;
  float4 z = make_float4(0.f, 0.f, 0.f, 0.f);
  for (; i < n4; i += stride) p[i] = z;
}

// ---------------------------------------------------------------------------
// K1: convert + transpose weights to f16: Wt[o*In + i] = (f16) W[i*Out + o]
// ---------------------------------------------------------------------------
__global__ void cvt_t(const float* __restrict__ W, _Float16* __restrict__ Wt,
                      int In, int Out) {
  int idx = blockIdx.x * blockDim.x + threadIdx.x;
  if (idx < In * Out) {
    int o = idx / In, i = idx - o * In;
    Wt[idx] = (_Float16)W[i * Out + o];
  }
}

// ---------------------------------------------------------------------------
// K2: self MLP  g_i = (relu(relu(x_i sW1 + sb1) sW2 + sb2)) sW3 + sb3
// ---------------------------------------------------------------------------
__global__ __launch_bounds__(256) void self_mlp(
    const float* __restrict__ x, const int* __restrict__ nbrs,
    const _Float16* __restrict__ Ws1, const _Float16* __restrict__ Ws2,
    const _Float16* __restrict__ ws3,
    const float* __restrict__ sb1, const float* __restrict__ sb2,
    const float* __restrict__ sb3,
    float* __restrict__ gbuf, float* __restrict__ selfw) {
  extern __shared__ char smem[];
  _Float16* As = (_Float16*)smem;                   // MBLK x LDA
  _Float16* Wb = As + MBLK * LDA;                   // 256 x LDB
  float*    Sred = (float*)(Wb + 256 * LDB);        // MBLK

  const int tid = threadIdx.x, lane = tid & 31, wave = tid >> 5;
  const int node0 = blockIdx.x * MBLK;

  // stage A: row r <- (f16) x[cur[node0+r]][0..127]
  {
    int r = tid >> 2, q = tid & 3;                  // 4 threads per row, 32 cols each
    int pos = node0 + r; if (pos >= NN) pos = NN - 1;
    int ci = nbrs[pos * (KNBR + 1)];                // cur (== pos by construction)
    const float* src = x + (size_t)ci * DD + q * 32;
    _Float16* dst = As + (size_t)r * LDA + q * 32;
    #pragma unroll
    for (int c = 0; c < 32; c += 4) {
      float4 f = *(const float4*)(src + c);
      v4h h; h[0] = (_Float16)f.x; h[1] = (_Float16)f.y;
      h[2] = (_Float16)f.z; h[3] = (_Float16)f.w;
      *(v4h*)(dst + c) = h;
    }
  }
  if (tid < MBLK) Sred[tid] = 0.f;
  __syncthreads();

  const int wm = wave >> 1, wn = wave & 1, m0 = wm * 16;
  v8f acc[8];
  #pragma unroll
  for (int t = 0; t < 8; ++t) acc[t] = zero8();

  gemm_layer<DD>(Ws1, As, Wb, m0, wn, lane, tid, acc);

  // h1 = relu(acc + sb1) -> As (f16), reset acc
  #pragma unroll
  for (int t = 0; t < 8; ++t) {
    int n = wn * 128 + t * 16 + (lane & 15);
    float bias = sb1[n];
    #pragma unroll
    for (int v = 0; v < 8; ++v) {
      int m = m0 + v + ((lane >> 4) << 3);
      float hv = acc[t][v] + bias;
      As[(size_t)m * LDA + n] = (_Float16)(hv > 0.f ? hv : 0.f);
      acc[t][v] = 0.f;
    }
  }
  __syncthreads();

  gemm_layer<HH>(Ws2, As, Wb, m0, wn, lane, tid, acc);

  // layer 3: dot(relu(acc + sb2), ws3)
  float partial[8];
  #pragma unroll
  for (int v = 0; v < 8; ++v) partial[v] = 0.f;
  #pragma unroll
  for (int t = 0; t < 8; ++t) {
    int n = wn * 128 + t * 16 + (lane & 15);
    float bias = sb2[n];
    float w3 = (float)ws3[n];
    #pragma unroll
    for (int v = 0; v < 8; ++v) {
      float hv = acc[t][v] + bias;
      hv = hv > 0.f ? hv : 0.f;
      partial[v] += hv * w3;
    }
  }
  #pragma unroll
  for (int v = 0; v < 8; ++v) {
    int m = m0 + v + ((lane >> 4) << 3);
    atomicAdd(&Sred[m], partial[v]);                // ds_add_f32
  }
  __syncthreads();

  if (tid < MBLK) {
    int pos = node0 + tid;
    if (pos < NN) {
      int ci = nbrs[pos * (KNBR + 1)];
      float val = Sred[tid] + sb3[0];
      gbuf[pos] = val;                              // g indexed by row position
      selfw[ci] = val;                              // self_w scattered at cur
    }
  }
}

// ---------------------------------------------------------------------------
// K3: neighbor MLP raw scores s_ij ; 2 nodes (64 pair-rows) per workgroup
// ---------------------------------------------------------------------------
__global__ __launch_bounds__(256) void pair_mlp(
    const float* __restrict__ x, const int* __restrict__ nbrs,
    const _Float16* __restrict__ Wn1, const _Float16* __restrict__ Wn2,
    const _Float16* __restrict__ wn3,
    const float* __restrict__ nb1, const float* __restrict__ nb2,
    const float* __restrict__ nb3,
    float* __restrict__ pairScore) {
  extern __shared__ char smem[];
  _Float16* As = (_Float16*)smem;
  _Float16* Wb = As + MBLK * LDA;
  float*    Sred = (float*)(Wb + 256 * LDB);

  const int tid = threadIdx.x, lane = tid & 31, wave = tid >> 5;
  const int node0 = blockIdx.x * 2;

  // stage A: row r = (node, j); cols 0..127 = x[cur], 128..255 = x[nbr]
  {
    int r = tid >> 2, q = tid & 3;                  // 4 threads/row, 64 cols each
    int pos = node0 + (r >> 5), j = r & 31;
    int ci  = nbrs[pos * (KNBR + 1)];
    int nb  = nbrs[pos * (KNBR + 1) + 1 + j];
    const float* src = (q < 2) ? (x + (size_t)ci * DD + q * 64)
                               : (x + (size_t)nb * DD + (q - 2) * 64);
    _Float16* dst = As + (size_t)r * LDA + q * 64;
    #pragma unroll
    for (int c = 0; c < 64; c += 4) {
      float4 f = *(const float4*)(src + c);
      v4h h; h[0] = (_Float16)f.x; h[1] = (_Float16)f.y;
      h[2] = (_Float16)f.z; h[3] = (_Float16)f.w;
      *(v4h*)(dst + c) = h;
    }
  }
  if (tid < MBLK) Sred[tid] = 0.f;
  __syncthreads();

  const int wm = wave >> 1, wn = wave & 1, m0 = wm * 16;
  v8f acc[8];
  #pragma unroll
  for (int t = 0; t < 8; ++t) acc[t] = zero8();

  gemm_layer<2 * DD>(Wn1, As, Wb, m0, wn, lane, tid, acc);

  #pragma unroll
  for (int t = 0; t < 8; ++t) {
    int n = wn * 128 + t * 16 + (lane & 15);
    float bias = nb1[n];
    #pragma unroll
    for (int v = 0; v < 8; ++v) {
      int m = m0 + v + ((lane >> 4) << 3);
      float hv = acc[t][v] + bias;
      As[(size_t)m * LDA + n] = (_Float16)(hv > 0.f ? hv : 0.f);
      acc[t][v] = 0.f;
    }
  }
  __syncthreads();

  gemm_layer<HH>(Wn2, As, Wb, m0, wn, lane, tid, acc);

  float partial[8];
  #pragma unroll
  for (int v = 0; v < 8; ++v) partial[v] = 0.f;
  #pragma unroll
  for (int t = 0; t < 8; ++t) {
    int n = wn * 128 + t * 16 + (lane & 15);
    float bias = nb2[n];
    float w3 = (float)wn3[n];
    #pragma unroll
    for (int v = 0; v < 8; ++v) {
      float hv = acc[t][v] + bias;
      hv = hv > 0.f ? hv : 0.f;
      partial[v] += hv * w3;
    }
  }
  #pragma unroll
  for (int v = 0; v < 8; ++v) {
    int m = m0 + v + ((lane >> 4) << 3);
    atomicAdd(&Sred[m], partial[v]);
  }
  __syncthreads();

  if (tid < MBLK) {
    int pos = node0 + (tid >> 5), j = tid & 31;
    pairScore[(size_t)pos * KNBR + j] = Sred[tid] + nb3[0];
  }
}

// ---------------------------------------------------------------------------
// K4: one wave per node: softmax(b*|s|), pw, scatter, Y_pred
// ---------------------------------------------------------------------------
__global__ __launch_bounds__(256) void finalize(
    const float* __restrict__ pairScore, const int* __restrict__ nbrs,
    const float* __restrict__ t, const float* __restrict__ eh,
    const float* __restrict__ g, const float* __restrict__ bptr,
    float* __restrict__ out) {
  const int wave = threadIdx.x >> 5, lane = threadIdx.x & 31;
  const int pos = blockIdx.x * 8 + wave;
  if (pos >= NN) return;
  const float b = bptr[0];
  const int ci = nbrs[pos * (KNBR + 1)];

  float s = pairScore[(size_t)pos * KNBR + lane];
  float a = b * fabsf(s);
  float m = a;
  #pragma unroll
  for (int off = 16; off > 0; off >>= 1) m = fmaxf(m, __shfl_xor(m, off, 32));
  float e = __expf(a - m);
  float sum = e;
  #pragma unroll
  for (int off = 16; off > 0; off >>= 1) sum += __shfl_xor(sum, off, 32);
  float pw = s * (e / sum);

  int nb = nbrs[pos * (KNBR + 1) + 1 + lane];
  float contrib = pw * (t[nb] - eh[nb]);
  #pragma unroll
  for (int off = 16; off > 0; off >>= 1) contrib += __shfl_xor(contrib, off, 32);

  // pairwise_w[ci][nb] = pw, with the post-pass diagonal zero folded in
  out[(size_t)NN + (size_t)ci * NN + nb] = (nb == ci) ? 0.f : pw;

  if (lane == 0)
    out[pos] = g[pos] * (t[ci] - eh[ci]) + contrib;
}

// ---------------------------------------------------------------------------
// host launcher
// ---------------------------------------------------------------------------
extern "C" void kernel_launch(void* const* d_in, const int* in_sizes, int n_in,
                              void* d_out, int out_size, void* d_ws, size_t ws_size,
                              hipStream_t stream) {
  const float* x    = (const float*)d_in[0];
  const int*   nbrs = (const int*)d_in[1];
  const float* t    = (const float*)d_in[2];
  const float* eh   = (const float*)d_in[3];
  const float* nW1  = (const float*)d_in[4];
  const float* nb1  = (const float*)d_in[5];
  const float* nW2  = (const float*)d_in[6];
  const float* nb2  = (const float*)d_in[7];
  const float* nW3  = (const float*)d_in[8];
  const float* nb3  = (const float*)d_in[9];
  const float* sW1  = (const float*)d_in[10];
  const float* sb1  = (const float*)d_in[11];
  const float* sW2  = (const float*)d_in[12];
  const float* sb2  = (const float*)d_in[13];
  const float* sW3  = (const float*)d_in[14];
  const float* sb3  = (const float*)d_in[15];
  const float* bsc  = (const float*)d_in[16];
  float* out = (float*)d_out;

  _Float16* wsh = (_Float16*)d_ws;
  float* pairScore = (float*)((char*)d_ws + HALF_BYTES);          // N*K f32
  float* gbuf      = pairScore + (size_t)NN * KNBR;               // N f32

  // K0: zero pairwise (N*N) + self_w (N) tail in one pass
  long long n4 = ((long long)NN * NN + NN) / 4;
  zero_pair<<<2048, 256, 0, stream>>>((float4*)(out + NN), n4);

  // K1: f32 -> transposed f16 weights
  cvt_t<<<(HH * 2 * DD + 255) / 256, 256, 0, stream>>>(nW1, wsh + OFF_N1, 2 * DD, HH);
  cvt_t<<<(HH * HH + 255) / 256, 256, 0, stream>>>(nW2, wsh + OFF_N2, HH, HH);
  cvt_t<<<1, 256, 0, stream>>>(nW3, wsh + OFF_N3, HH, 1);
  cvt_t<<<(DD * HH + 255) / 256, 256, 0, stream>>>(sW1, wsh + OFF_S1, DD, HH);
  cvt_t<<<(HH * HH + 255) / 256, 256, 0, stream>>>(sW2, wsh + OFF_S2, HH, HH);
  cvt_t<<<1, 256, 0, stream>>>(sW3, wsh + OFF_S3, HH, 1);

  // K2: self MLP
  self_mlp<<<(NN + MBLK - 1) / MBLK, 256, SMEM_BYTES, stream>>>(
      x, nbrs, wsh + OFF_S1, wsh + OFF_S2, wsh + OFF_S3,
      sb1, sb2, sb3, gbuf, out + NN + (size_t)NN * NN);

  // K3: neighbor MLP (2 nodes per WG)
  pair_mlp<<<NN / 2, 256, SMEM_BYTES, stream>>>(
      x, nbrs, wsh + OFF_N1, wsh + OFF_N2, wsh + OFF_N3,
      nb1, nb2, nb3, pairScore);

  // K4: softmax + scatter + Y_pred (8 nodes per block, 1 wave each)
  finalize<<<(NN + 7) / 8, 256, 0, stream>>>(pairScore, nbrs, t, eh, gbuf, bsc, out);
}
